// DeepSeekBlock_18425409700064
// MI455X (gfx1250) — matrix operand order repaired and checked by reading, not compiled, not run on hardware
//
#include <hip/hip_runtime.h>

// ---------------------------------------------------------------------------
// DeepSeek block for MI455X (gfx1250). Compute-bound (~290 GFLOP vs ~15us of
// HBM traffic at 23.3 TB/s) => everything matrix-shaped goes through
// v_wmma_f32_16x16x32_bf16 (f32 accumulate). Weights pre-converted once per
// launch to bf16-transposed layout so every WMMA operand fragment is a
// contiguous 16B/32B vector load. GEMM B tiles are staged into LDS by the
// Tensor Data Mover (tensor_load_to_lds, double-buffered on TENSORcnt).
// ---------------------------------------------------------------------------

typedef __attribute__((ext_vector_type(16))) __bf16 v16bf;
typedef __attribute__((ext_vector_type(8)))  __bf16 v8bf;
typedef __attribute__((ext_vector_type(8)))  float  v8f;
typedef __attribute__((ext_vector_type(4)))  unsigned int u32x4;
typedef __attribute__((ext_vector_type(8)))  int          i32x8;
typedef __attribute__((ext_vector_type(4)))  int          i32x4;
typedef unsigned int u32;

namespace cfg {
constexpr int B = 2, L = 2048, D = 1024, H = 16, HD = 64, R = 256, E = 8,
              F = 1024, T = B * L;
}

// D = A*B + C, bf16 inputs, f32 accum
__device__ inline v8f wmma_bf16(v16bf a, v16bf b, v8f c) {
  return __builtin_amdgcn_wmma_f32_16x16x32_bf16(
      /*neg_a=*/false, a, /*neg_b=*/false, b, /*c_mod=*/(short)0, c,
      /*reuse_a=*/false, /*reuse_b=*/false);
}

// ---- TDM: 2D bf16 tile (tile_k contiguous elems x tile_n rows) -> LDS ------
// D# per CDNA5 ISA §8.3/8.4. Groups 2/3 zero (<=2D tensor). cpol=0.
// 6-arg builtin form (clang-23 / amdgpu-toolchain).
__device__ inline void tdm_load_2d_bf16(u32 lds_off, const void* gaddr,
                                        u32 tile_k, u32 tile_n,
                                        u32 stride_elems, u32 tensor_k) {
  const u32 lo = (u32)(size_t)gaddr;
  const u32 hi = (u32)((size_t)gaddr >> 32);
  u32x4 g0;
  g0[0] = 1u;                               // count=1 (valid user descriptor)
  g0[1] = lds_off;                          // LDS byte address
  g0[2] = lo;                               // global_addr[31:0]
  g0[3] = (hi & 0x01FFFFFFu) | (2u << 30);  // global_addr[56:32] | type=2
  i32x8 g1;
  g1[0] = (int)(1u << 16);                  // data_size=1 (2 bytes)
  g1[1] = (int)((tensor_k & 0xFFFFu) << 16);          // tensor_dim0[15:0]
  g1[2] = (int)((tensor_k >> 16) | (0xFFFFu << 16));  // dim0[31:16]|dim1[15:0]
  g1[3] = (int)((tile_k & 0xFFFFu) << 16);  // tensor_dim1[31:16]=0 | tile_dim0
  g1[4] = (int)(tile_n & 0xFFFFu);          // tile_dim1 | tile_dim2=0
  g1[5] = (int)stride_elems;                // tensor_dim0_stride[31:0]
  g1[6] = 0;                                // stride[47:32] | dim1_stride lo
  g1[7] = 0;
  const i32x4 z4 = {0, 0, 0, 0};
  const i32x8 z8 = {0, 0, 0, 0, 0, 0, 0, 0};
  __builtin_amdgcn_tensor_load_to_lds(g0, g1, z4, z4, z8, 0);
}

// ---- fragment loaders (contiguous k order, same for A and B) ---------------------------
// A (16x32, 16-bit): lane m = lane&15; lanes 0-15 hold K {0..15},
// lanes 16-31 hold K {16..31}. Per lane: two contiguous 16B chunks.
__device__ inline v16bf load_a_bf16(const __bf16* __restrict__ A, int lda,
                                    size_t row0, int k0, int lane) {
  const int m = lane & 15;
  const int koff = (lane < 16) ? 0 : 16;
  const __bf16* p = A + (row0 + (size_t)m) * (size_t)lda + k0 + koff;
  v8bf lo = *(const v8bf*)p;         // K = koff .. koff+7
  v8bf hi = *(const v8bf*)(p + 8);   // K = koff+8 .. koff+15
  v16bf a;
#pragma unroll
  for (int j = 0; j < 8; ++j) {
    a[j] = lo[j];
    a[8 + j] = hi[j];
  }
  return a;
}

// B (32x16, 16-bit): lane n = lane&15 is the column; lanes 0-15 cover
// K=0..15, lanes 16-31 cover K=16..31. With B stored transposed ([N,K]
// row-major) a lane's fragment is 16 contiguous bf16 = one 32B load.
__device__ inline v16bf load_bT(const __bf16* __restrict__ BT, int ldk,
                                int col0, int k0, int lane) {
  const int n = lane & 15;
  const int koff = (lane < 16) ? 0 : 16;
  return *(const v16bf*)(BT + (size_t)(col0 + n) * (size_t)ldk + k0 + koff);
}

// K^T fragment for S = Q x K^T: element (kk,n) = K[key0+n][kcol0+kk]
__device__ inline v16bf load_b_kT(const __bf16* __restrict__ Kb, int ld,
                                  size_t key0, int kcol0, int lane) {
  const int n = lane & 15;
  const int koff = (lane < 16) ? 0 : 16;
  return *(const v16bf*)(Kb + (key0 + (size_t)n) * (size_t)ld + kcol0 + koff);
}

// C/D (16x16 f32): VGPR r -> lanes 0-15: (M=r, N=lane); lanes 16-31: (M=r+8).
__device__ inline void store_tile(float* Cf, __bf16* Cbf, int ldc, size_t row0,
                                  int col0, const v8f c, int lane,
                                  const float* __restrict__ rowscale,
                                  int rs_stride, int acc) {
  const int n = lane & 15;
  const int mb = (lane < 16) ? 0 : 8;
  const int col = col0 + n;
#pragma unroll
  for (int r = 0; r < 8; ++r) {
    const size_t row = row0 + mb + r;
    float v = c[r];
    if (rowscale) v *= rowscale[row * (size_t)rs_stride];
    const size_t idx = row * (size_t)ldc + col;
    if (Cf) {
      if (acc) Cf[idx] += v; else Cf[idx] = v;
    }
    if (Cbf) Cbf[idx] = (__bf16)v;
  }
}

// ---------------------------------------------------------------------------
// GEMM: C[M,N] (+)= rowscale[m] * (A[M,K] @ B[K,N]); A bf16 [M,K], B given as
// bf16 transposed BT [N,K]. 4 waves/block, wave owns a 32x32 C tile. The
// block's 64x32 BT tile is staged into LDS by TDM, double-buffered.
// Requires M%64==0, N%64==0, K%32==0 (true for all launches here).
// ---------------------------------------------------------------------------
__global__ __launch_bounds__(128) void gemm_bf16_kernel(
    const __bf16* __restrict__ A, int lda, const __bf16* __restrict__ BT,
    float* Cf, __bf16* Cbf, int ldc, int M, int N, int Kd,
    const float* __restrict__ rowscale, int rs_stride, int acc) {
  const int lane = threadIdx.x & 31;
  const int wave = threadIdx.x >> 5;
  const size_t bm = (size_t)blockIdx.y * 64 + (size_t)(wave >> 1) * 32;
  const int bn_blk = blockIdx.x * 64;
  const int bn_loc = (wave & 1) * 32;

  __shared__ __attribute__((aligned(64))) __bf16 Bbuf[2][64 * 32];
  const u32 lds0 = (u32)(size_t)(&Bbuf[0][0]);
  const u32 lds1 = (u32)(size_t)(&Bbuf[1][0]);
  const __bf16* Bg = BT + (size_t)bn_blk * Kd;  // block's 64 BT rows

  const int nk = Kd / 32;
  if (wave == 0) tdm_load_2d_bf16(lds0, Bg, 32, 64, (u32)Kd, (u32)Kd);

  v8f c00 = {}, c01 = {}, c10 = {}, c11 = {};
  for (int kt = 0; kt < nk; ++kt) {
    const int k0 = kt * 32;
    if (wave == 0) {
      if (kt + 1 < nk) {
        tdm_load_2d_bf16((kt & 1) ? lds0 : lds1, Bg + (k0 + 32), 32, 64,
                         (u32)Kd, (u32)Kd);
        __builtin_amdgcn_s_wait_tensorcnt(1);  // current buffer complete
      } else {
        __builtin_amdgcn_s_wait_tensorcnt(0);
      }
    }
    __syncthreads();  // publish LDS tile to all waves

    const __bf16* Bl = &Bbuf[kt & 1][0];  // [64 cols][32 k] row-major
    v16bf a0 = load_a_bf16(A, lda, bm, k0, lane);
    v16bf a1 = load_a_bf16(A, lda, bm + 16, k0, lane);
    v16bf b0 = load_bT(Bl, 32, bn_loc, 0, lane);
    v16bf b1 = load_bT(Bl, 32, bn_loc + 16, 0, lane);
    if (k0 + 32 < Kd)  // prefetch next A lines
      __builtin_prefetch(A + (bm + (lane & 15)) * (size_t)lda + k0 + 32, 0, 1);
    c00 = wmma_bf16(a0, b0, c00);
    c01 = wmma_bf16(a0, b1, c01);
    c10 = wmma_bf16(a1, b0, c10);
    c11 = wmma_bf16(a1, b1, c11);
    __syncthreads();  // buffer reuse
  }
  const int bn = bn_blk + bn_loc;
  store_tile(Cf, Cbf, ldc, bm,      bn,      c00, lane, rowscale, rs_stride, acc);
  store_tile(Cf, Cbf, ldc, bm,      bn + 16, c01, lane, rowscale, rs_stride, acc);
  store_tile(Cf, Cbf, ldc, bm + 16, bn,      c10, lane, rowscale, rs_stride, acc);
  store_tile(Cf, Cbf, ldc, bm + 16, bn + 16, c11, lane, rowscale, rs_stride, acc);
}

// ---------------------------------------------------------------------------
// Weight convert + transpose: W[K,N] f32 -> WT[N,K] bf16 (LDS-tiled, 32x32).
// ---------------------------------------------------------------------------
__global__ __launch_bounds__(256) void wt_convert_kernel(
    const float* __restrict__ W, __bf16* __restrict__ WT, int K, int N) {
  __shared__ float tile[32][33];
  const int k0 = blockIdx.x * 32, n0 = blockIdx.y * 32;
  const int tx = threadIdx.x & 31, ty = threadIdx.x >> 5;  // 32x8
#pragma unroll
  for (int i = ty; i < 32; i += 8)
    tile[i][tx] = W[(size_t)(k0 + i) * N + (n0 + tx)];
  __syncthreads();
#pragma unroll
  for (int i = ty; i < 32; i += 8)
    WT[(size_t)(n0 + i) * K + (k0 + tx)] = (__bf16)tile[tx][i];
}

// bf16 [L,C] (strided inside kv) -> bf16 [C,L] per batch (for V^T in PV)
__global__ __launch_bounds__(256) void transpose_v_kernel(
    const __bf16* __restrict__ kv, __bf16* __restrict__ VT, int Lq, int C,
    int ldkv) {
  __shared__ __bf16 tile[32][33];
  const int l0 = blockIdx.x * 32, c0 = blockIdx.y * 32, b = blockIdx.z;
  const int tx = threadIdx.x & 31, ty = threadIdx.x >> 5;
  const __bf16* src = kv + ((size_t)b * Lq) * ldkv + C;  // v section
#pragma unroll
  for (int i = ty; i < 32; i += 8)
    tile[i][tx] = src[(size_t)(l0 + i) * ldkv + (c0 + tx)];
  __syncthreads();
  __bf16* dst = VT + ((size_t)b * C) * Lq;
#pragma unroll
  for (int i = ty; i < 32; i += 8)
    dst[(size_t)(c0 + i) * Lq + (l0 + tx)] = tile[tx][i];
}

// ---------------------------------------------------------------------------
// Flash attention (causal, online softmax). One wave per 16-query tile/head.
// QK^T and PV via WMMA; P relaid C->A layout through LDS.
// ---------------------------------------------------------------------------
__global__ __launch_bounds__(32) void mla_attn_kernel(
    const __bf16* __restrict__ Q,   // [T, H*HD] bf16
    const __bf16* __restrict__ KV,  // [T, 2*H*HD] bf16 (k cols first)
    const __bf16* __restrict__ VT,  // [B][H*HD][L] bf16 (v transposed)
    __bf16* __restrict__ O,         // [T, H*HD] bf16
    int Lq, int Hn, int HDn) {
  const int lane = threadIdx.x & 31;
  const int qi0 = blockIdx.x * 16;
  const int h = blockIdx.y;
  const int b = blockIdx.z;
  const int ldq = Hn * HDn;       // 1024
  const int ldkv = 2 * Hn * HDn;  // 2048
  const size_t rowbase = (size_t)b * Lq;
  const int mbase = (lane < 16) ? 0 : 8;
  const int ncol = lane & 15;
  const int koffB = (lane < 16) ? 0 : 16;

  const __bf16* Qh = Q + (size_t)h * HDn;
  const __bf16* Kh = KV + (size_t)h * HDn;
  const __bf16* VTb = VT + ((size_t)b * ldq) * Lq + (size_t)h * HDn * Lq;

  v16bf qf0 = load_a_bf16(Qh, ldq, rowbase + qi0, 0, lane);
  v16bf qf1 = load_a_bf16(Qh, ldq, rowbase + qi0, 32, lane);

  v8f o0 = {}, o1 = {}, o2 = {}, o3 = {};
  float mrow[8], lrow[8];
#pragma unroll
  for (int r = 0; r < 8; ++r) { mrow[r] = -3.0e38f; lrow[r] = 0.f; }

  __shared__ __attribute__((aligned(32))) __bf16 Plds[16 * 32];
  const int nchunk = (qi0 + 16 + 31) / 32;

  for (int c = 0; c < nchunk; ++c) {
    const int key0 = c * 32;
    v8f s0 = {}, s1 = {};
    s0 = wmma_bf16(qf0, load_b_kT(Kh, ldkv, rowbase + key0, 0, lane), s0);
    s0 = wmma_bf16(qf1, load_b_kT(Kh, ldkv, rowbase + key0, 32, lane), s0);
    s1 = wmma_bf16(qf0, load_b_kT(Kh, ldkv, rowbase + key0 + 16, 0, lane), s1);
    s1 = wmma_bf16(qf1, load_b_kT(Kh, ldkv, rowbase + key0 + 16, 32, lane), s1);

    // scale (1/sqrt(64)) + causal mask in C layout
#pragma unroll
    for (int r = 0; r < 8; ++r) {
      const int qrow = qi0 + mbase + r;
      float a = s0[r] * 0.125f;
      float bb = s1[r] * 0.125f;
      if (key0 + ncol > qrow) a = -3.0e38f;
      if (key0 + 16 + ncol > qrow) bb = -3.0e38f;
      s0[r] = a;
      s1[r] = bb;
    }

    // online softmax (rows live across 16-lane halves; shfl-reduce)
#pragma unroll
    for (int r = 0; r < 8; ++r) {
      float rm = fmaxf(s0[r], s1[r]);
#pragma unroll
      for (int off = 8; off; off >>= 1) rm = fmaxf(rm, __shfl_xor(rm, off));
      const float mnew = fmaxf(mrow[r], rm);
      const float sc = __expf(mrow[r] - mnew);
      const float p0 = __expf(s0[r] - mnew);
      const float p1 = __expf(s1[r] - mnew);
      float rs = p0 + p1;
#pragma unroll
      for (int off = 8; off; off >>= 1) rs += __shfl_xor(rs, off);
      lrow[r] = lrow[r] * sc + rs;
      mrow[r] = mnew;
      o0[r] *= sc; o1[r] *= sc; o2[r] *= sc; o3[r] *= sc;
      Plds[(mbase + r) * 32 + ncol] = (__bf16)p0;
      Plds[(mbase + r) * 32 + 16 + ncol] = (__bf16)p1;
    }
    __syncthreads();  // single wave: DS-count ordering

    v16bf pf = load_a_bf16(&Plds[0], 32, 0, 0, lane);
    // V^T fragments: 32B contiguous per lane
    v16bf vb0 = *(const v16bf*)(VTb + (size_t)(0 + ncol) * Lq + key0 + koffB);
    v16bf vb1 = *(const v16bf*)(VTb + (size_t)(16 + ncol) * Lq + key0 + koffB);
    v16bf vb2 = *(const v16bf*)(VTb + (size_t)(32 + ncol) * Lq + key0 + koffB);
    v16bf vb3 = *(const v16bf*)(VTb + (size_t)(48 + ncol) * Lq + key0 + koffB);
    o0 = wmma_bf16(pf, vb0, o0);
    o1 = wmma_bf16(pf, vb1, o1);
    o2 = wmma_bf16(pf, vb2, o2);
    o3 = wmma_bf16(pf, vb3, o3);
    __syncthreads();
  }

  __bf16* Oh = O + (size_t)h * HDn;
  const size_t row0 = rowbase + qi0;
#pragma unroll
  for (int r = 0; r < 8; ++r) {
    const float inv = 1.f / lrow[r];
    const size_t row = row0 + mbase + r;
    Oh[row * ldq + 0 + ncol]  = (__bf16)(o0[r] * inv);
    Oh[row * ldq + 16 + ncol] = (__bf16)(o1[r] * inv);
    Oh[row * ldq + 32 + ncol] = (__bf16)(o2[r] * inv);
    Oh[row * ldq + 48 + ncol] = (__bf16)(o3[r] * inv);
  }
}

// ---------------------------------------------------------------------------
// Elementwise / small kernels
// ---------------------------------------------------------------------------
__global__ void rmsnorm_kernel(const float* __restrict__ x,
                               const float* __restrict__ w, __bf16* out_bf,
                               float* out_f, int Dm) {
  const int row = blockIdx.x;
  const float* xr = x + (size_t)row * Dm;
  __shared__ float red[8];
  float ss = 0.f;
  for (int i = threadIdx.x; i < Dm; i += blockDim.x) {
    float v = xr[i];
    ss += v * v;
  }
#pragma unroll
  for (int off = 16; off; off >>= 1) ss += __shfl_xor(ss, off);
  if ((threadIdx.x & 31) == 0) red[threadIdx.x >> 5] = ss;
  __syncthreads();
  float s = 0.f;
  const int nw = blockDim.x >> 5;
  for (int i = 0; i < nw; ++i) s += red[i];
  const float inv = rsqrtf(s / (float)Dm + 1e-6f);
  for (int i = threadIdx.x; i < Dm; i += blockDim.x) {
    const float v = xr[i] * inv * w[i];
    if (out_bf) out_bf[(size_t)row * Dm + i] = (__bf16)v;
    if (out_f) out_f[(size_t)row * Dm + i] = v;
  }
}

// logits[t, 0..7] = hn[t,:] . gate_w[:, 0..7]   (N=8 too small for WMMA tiles)
__global__ __launch_bounds__(256) void gate_logits_kernel(
    const __bf16* __restrict__ hn, const float* __restrict__ gate_w,
    float* __restrict__ logits) {
  const int t = blockIdx.x;
  __shared__ float red[8][cfg::E];
  float acc[cfg::E];
#pragma unroll
  for (int e = 0; e < cfg::E; ++e) acc[e] = 0.f;
  for (int k = threadIdx.x; k < cfg::D; k += blockDim.x) {
    const float xv = (float)hn[(size_t)t * cfg::D + k];
#pragma unroll
    for (int e = 0; e < cfg::E; ++e) acc[e] += xv * gate_w[k * cfg::E + e];
  }
#pragma unroll
  for (int e = 0; e < cfg::E; ++e) {
#pragma unroll
    for (int off = 16; off; off >>= 1) acc[e] += __shfl_xor(acc[e], off);
  }
  if ((threadIdx.x & 31) == 0) {
#pragma unroll
    for (int e = 0; e < cfg::E; ++e) red[threadIdx.x >> 5][e] = acc[e];
  }
  __syncthreads();
  if (threadIdx.x < cfg::E) {
    float s = 0.f;
    for (int wv = 0; wv < 8; ++wv) s += red[wv][threadIdx.x];
    logits[(size_t)t * cfg::E + threadIdx.x] = s;
  }
}

__global__ void router_kernel(const float* __restrict__ logits,
                              float* __restrict__ probs_out,
                              float* __restrict__ combine, int T) {
  const int t = blockIdx.x * blockDim.x + threadIdx.x;
  if (t >= T) return;
  float l[cfg::E];
  float mx = -3.0e38f;
#pragma unroll
  for (int e = 0; e < cfg::E; ++e) {
    l[e] = logits[(size_t)t * cfg::E + e];
    mx = fmaxf(mx, l[e]);
  }
  float s = 0.f;
#pragma unroll
  for (int e = 0; e < cfg::E; ++e) {
    l[e] = __expf(l[e] - mx);
    s += l[e];
  }
  const float inv = 1.f / s;
  float p[cfg::E];
#pragma unroll
  for (int e = 0; e < cfg::E; ++e) {
    p[e] = l[e] * inv;
    probs_out[(size_t)t * cfg::E + e] = p[e];
  }
  int i0 = 0; float b0 = p[0];
#pragma unroll
  for (int e = 1; e < cfg::E; ++e)
    if (p[e] > b0) { b0 = p[e]; i0 = e; }
  int i1 = -1; float b1 = -1.f;
#pragma unroll
  for (int e = 0; e < cfg::E; ++e)
    if (e != i0 && p[e] > b1) { b1 = p[e]; i1 = e; }
  const float wsum = b0 + b1;
#pragma unroll
  for (int e = 0; e < cfg::E; ++e) combine[(size_t)t * cfg::E + e] = 0.f;
  combine[(size_t)t * cfg::E + i0] = b0 / wsum;
  combine[(size_t)t * cfg::E + i1] = b1 / wsum;
}

__global__ void residual_add_kernel(const float* __restrict__ x, float* h,
                                    size_t n) {
  const size_t i = (size_t)blockIdx.x * blockDim.x + threadIdx.x;
  if (i < n) h[i] = x[i] + h[i];
}

__global__ void silu_mul_kernel(const float* __restrict__ h1,
                                const float* __restrict__ h3, __bf16* act,
                                size_t n) {
  const size_t i = (size_t)blockIdx.x * blockDim.x + threadIdx.x;
  if (i < n) {
    const float a = h1[i];
    const float s = a / (1.f + __expf(-a));
    act[i] = (__bf16)(s * h3[i]);
  }
}

__global__ void final_add_kernel(const float* __restrict__ h,
                                 const float* __restrict__ moe,
                                 float* __restrict__ out, size_t n) {
  const size_t i = (size_t)blockIdx.x * blockDim.x + threadIdx.x;
  if (i < n) out[i] = h[i] + moe[i];
}

// ---------------------------------------------------------------------------
// Host orchestration
// ---------------------------------------------------------------------------
extern "C" void kernel_launch(void* const* d_in, const int* in_sizes, int n_in,
                              void* d_out, int out_size, void* d_ws,
                              size_t ws_size, hipStream_t stream) {
  (void)in_sizes; (void)n_in; (void)out_size; (void)ws_size;
  using namespace cfg;

  const float* x      = (const float*)d_in[0];
  const float* ln1_w  = (const float*)d_in[1];
  const float* q_w    = (const float*)d_in[2];
  const float* kvd_w  = (const float*)d_in[3];
  const float* kvn_w  = (const float*)d_in[4];
  const float* kvu_w  = (const float*)d_in[5];
  const float* o_w    = (const float*)d_in[6];
  const float* ln2_w  = (const float*)d_in[7];
  const float* gate_w = (const float*)d_in[8];
  const float* sh_w1  = (const float*)d_in[9];
  const float* sh_w2  = (const float*)d_in[10];
  const float* sh_w3  = (const float*)d_in[11];
  const float* re_w1  = (const float*)d_in[12];
  const float* re_w2  = (const float*)d_in[13];
  const float* re_w3  = (const float*)d_in[14];

  float* out_y = (float*)d_out;              // [T, D]
  float* out_probs = out_y + (size_t)T * D;  // [T, E]

  // workspace carve-out (~152 MiB)
  size_t off = 0;
  auto take = [&](size_t bytes) -> void* {
    void* p = (char*)d_ws + off;
    off += (bytes + 255) & ~(size_t)255;
    return p;
  };
  __bf16* xn_bf     = (__bf16*)take((size_t)T * D * 2);
  __bf16* q_bf      = (__bf16*)take((size_t)T * D * 2);
  float*  latent_f  = (float*) take((size_t)T * R * 4);
  __bf16* latent_bf = (__bf16*)take((size_t)T * R * 2);
  __bf16* kv_bf     = (__bf16*)take((size_t)T * 2 * D * 2);
  __bf16* vT_bf     = (__bf16*)take((size_t)T * D * 2);
  __bf16* attn_bf   = (__bf16*)take((size_t)T * D * 2);
  float*  h_f       = (float*) take((size_t)T * D * 4);
  __bf16* hn_bf     = (__bf16*)take((size_t)T * D * 2);
  float*  logits_f  = (float*) take((size_t)T * E * 4);
  float*  combine_f = (float*) take((size_t)T * E * 4);
  float*  h1_f      = (float*) take((size_t)T * F * 4);
  float*  h3_f      = (float*) take((size_t)T * F * 4);
  __bf16* act_bf    = (__bf16*)take((size_t)T * F * 2);
  float*  moe_f     = (float*) take((size_t)T * D * 4);
  // bf16-transposed weights
  __bf16* qT   = (__bf16*)take((size_t)D * H * HD * 2);
  __bf16* kvdT = (__bf16*)take((size_t)D * R * 2);
  __bf16* kvuT = (__bf16*)take((size_t)R * 2 * H * HD * 2);
  __bf16* oT   = (__bf16*)take((size_t)H * HD * D * 2);
  __bf16* sh1T = (__bf16*)take((size_t)D * F * 2);
  __bf16* sh2T = (__bf16*)take((size_t)F * D * 2);
  __bf16* sh3T = (__bf16*)take((size_t)D * F * 2);
  __bf16* e1T  = (__bf16*)take((size_t)D * F * 2);  // per-expert staging
  __bf16* e2T  = (__bf16*)take((size_t)F * D * 2);
  __bf16* e3T  = (__bf16*)take((size_t)D * F * 2);

  auto wt = [&](const float* W, __bf16* WT, int K, int N) {
    wt_convert_kernel<<<dim3(K / 32, N / 32), 256, 0, stream>>>(W, WT, K, N);
  };
  auto gemm = [&](const __bf16* A, int lda, const __bf16* BT, float* Cf,
                  __bf16* Cbf, int ldc, int M, int N, int Kd,
                  const float* rowscale, int rs_stride, int acc) {
    dim3 g(N / 64, M / 64);
    gemm_bf16_kernel<<<g, 128, 0, stream>>>(A, lda, BT, Cf, Cbf, ldc, M, N, Kd,
                                            rowscale, rs_stride, acc);
  };

  const size_t nTD = (size_t)T * D;
  const size_t nTF = (size_t)T * F;
  const int EW = 256;
  const int gTD = (int)((nTD + EW - 1) / EW);
  const int gTF = (int)((nTF + EW - 1) / EW);

  // ---- weight conversion (bf16, transposed) ----
  wt(q_w, qT, D, H * HD);
  wt(kvd_w, kvdT, D, R);
  wt(kvu_w, kvuT, R, 2 * H * HD);
  wt(o_w, oT, H * HD, D);
  wt(sh_w1, sh1T, D, F);
  wt(sh_w2, sh2T, F, D);
  wt(sh_w3, sh3T, D, F);

  // ---- attention ----
  rmsnorm_kernel<<<T, 256, 0, stream>>>(x, ln1_w, xn_bf, nullptr, D);
  gemm(xn_bf, D, qT, nullptr, q_bf, H * HD, T, H * HD, D, nullptr, 0, 0);
  gemm(xn_bf, D, kvdT, latent_f, nullptr, R, T, R, D, nullptr, 0, 0);
  rmsnorm_kernel<<<T, 256, 0, stream>>>(latent_f, kvn_w, latent_bf, nullptr, R);
  gemm(latent_bf, R, kvuT, nullptr, kv_bf, 2 * H * HD, T, 2 * H * HD, R,
       nullptr, 0, 0);
  transpose_v_kernel<<<dim3(L / 32, (H * HD) / 32, B), 256, 0, stream>>>(
      kv_bf, vT_bf, L, H * HD, 2 * H * HD);
  mla_attn_kernel<<<dim3(L / 16, H, B), 32, 0, stream>>>(q_bf, kv_bf, vT_bf,
                                                         attn_bf, L, H, HD);
  gemm(attn_bf, H * HD, oT, h_f, nullptr, D, T, D, H * HD, nullptr, 0, 0);
  residual_add_kernel<<<gTD, EW, 0, stream>>>(x, h_f, nTD);

  // ---- MoE ----
  rmsnorm_kernel<<<T, 256, 0, stream>>>(h_f, ln2_w, hn_bf, nullptr, D);
  gate_logits_kernel<<<T, 256, 0, stream>>>(hn_bf, gate_w, logits_f);
  router_kernel<<<(T + EW - 1) / EW, EW, 0, stream>>>(logits_f, out_probs,
                                                      combine_f, T);
  // shared expert (initializes moe_f)
  gemm(hn_bf, D, sh1T, h1_f, nullptr, F, T, F, D, nullptr, 0, 0);
  gemm(hn_bf, D, sh3T, h3_f, nullptr, F, T, F, D, nullptr, 0, 0);
  silu_mul_kernel<<<gTF, EW, 0, stream>>>(h1_f, h3_f, act_bf, nTF);
  gemm(act_bf, F, sh2T, moe_f, nullptr, D, T, D, F, nullptr, 0, 0);
  // routed experts: combine weight folded into the w2 GEMM epilogue
  for (int e = 0; e < E; ++e) {
    wt(re_w1 + (size_t)e * D * F, e1T, D, F);
    wt(re_w3 + (size_t)e * D * F, e3T, D, F);
    wt(re_w2 + (size_t)e * F * D, e2T, F, D);
    gemm(hn_bf, D, e1T, h1_f, nullptr, F, T, F, D, nullptr, 0, 0);
    gemm(hn_bf, D, e3T, h3_f, nullptr, F, T, F, D, nullptr, 0, 0);
    silu_mul_kernel<<<gTF, EW, 0, stream>>>(h1_f, h3_f, act_bf, nTF);
    gemm(act_bf, F, e2T, moe_f, nullptr, D, T, D, F, combine_f + e, E, 1);
  }

  final_add_kernel<<<gTD, EW, 0, stream>>>(h_f, moe_f, out_y, nTD);
}
